// Aggregator_89258010346031
// MI455X (gfx1250) — compile-verified, matrix operand order checked by reading, not run on hardware
//
#include <hip/hip_runtime.h>
#include <math.h>

#define N_ENT   100000
#define N_USERS 30000
#define N_ITEMS 20000
#define DEMB    64
#define EPSV    1e-5f
#define MINV    1e-15f

typedef __attribute__((ext_vector_type(16))) _Float16 v16h;
typedef __attribute__((ext_vector_type(8)))  float    v8f;

// ---------------- wave32 helpers ----------------
__device__ __forceinline__ float wsum(float v) {
  v += __shfl_xor(v, 16, 32);
  v += __shfl_xor(v, 8, 32);
  v += __shfl_xor(v, 4, 32);
  v += __shfl_xor(v, 2, 32);
  v += __shfl_xor(v, 1, 32);
  return v;
}

// mobius_add with analytic squared-norm of the result.
// x,y are per-lane float2 components; p2,q2,pq are wave-uniform scalars.
struct M2 { float x, y, n2; };
__device__ __forceinline__ M2 mobius_add2(float px, float py, float p2,
                                          float qx, float qy, float q2,
                                          float pq) {
  float a   = 1.0f + 2.0f * pq + q2;
  float b   = 1.0f - p2;
  float den = fmaxf(1.0f + 2.0f * pq + p2 * q2, MINV);
  float inv = 1.0f / den;
  M2 r;
  r.x  = (a * px + b * qx) * inv;
  r.y  = (a * py + b * qy) * inv;
  r.n2 = (a * a * p2 + 2.0f * a * b * pq + b * b * q2) * inv * inv;
  return r;
}

// project() onto the Poincare ball; n2 is tracked analytically (wave-uniform).
__device__ __forceinline__ void project2(float& x, float& y, float& n2) {
  float n = sqrtf(fmaxf(n2, MINV));
  const float m = 1.0f - EPSV;
  if (n > m) {            // uniform branch (n is wave-uniform)
    float s = m / n;
    x *= s; y *= s; n2 *= s * s;
  }
}

// ---------------- KG hyperbolic edge kernel (1 wave / edge) ----------------
__global__ void kg_edge_kernel(const float* __restrict__ ent,
                               const float* __restrict__ relt,
                               const int* __restrict__ src,
                               const int* __restrict__ dst,
                               const int* __restrict__ typ,
                               float* __restrict__ sum_kg,
                               float* __restrict__ cnt_kg,
                               int n_edges) {
  int gid  = blockIdx.x * blockDim.x + threadIdx.x;
  int wave = gid >> 5;
  int lane = threadIdx.x & 31;
  if (wave >= n_edges) return;

  int s = src[wave];
  int d = dst[wave];
  int t = typ[wave] + 2;
  int d0 = lane * 2;

  float2 xs = *(const float2*)(ent  + (size_t)s * DEMB + d0);
  float2 xd = *(const float2*)(ent  + (size_t)d * DEMB + d0);
  float2 xr = *(const float2*)(relt + (size_t)t * DEMB + d0);

  // ---- p = expmap0(xs) = project(tanh(n) * xs / n) ----
  float u2 = wsum(xs.x * xs.x + xs.y * xs.y);                 // reduction 1
  float un = sqrtf(fmaxf(u2, MINV));
  float sc = tanhf(un) / un;
  float px = sc * xs.x, py = sc * xs.y;
  float p2 = sc * sc * u2;
  project2(px, py, p2);

  float lam_p = 2.0f / fmaxf(1.0f - p2, MINV);

  // ---- dst_h = expmap(xd, p) ----
  float v2 = wsum(xd.x * xd.x + xd.y * xd.y);                 // reduction 2
  float vn = sqrtf(fmaxf(v2, MINV));
  float sv = tanhf(0.5f * lam_p * vn) / vn;
  float pq = sv * wsum(px * xd.x + py * xd.y);                // reduction 3
  M2 dh = mobius_add2(px, py, p2, sv * xd.x, sv * xd.y, sv * sv * v2, pq);
  project2(dh.x, dh.y, dh.n2);

  // ---- rel_h = expmap(xr, p) ----
  float w2 = wsum(xr.x * xr.x + xr.y * xr.y);                 // reduction 4
  float wn = sqrtf(fmaxf(w2, MINV));
  float sw = tanhf(0.5f * lam_p * wn) / wn;
  float pr = sw * wsum(px * xr.x + py * xr.y);                // reduction 5
  M2 rh = mobius_add2(px, py, p2, sw * xr.x, sw * xr.y, sw * sw * w2, pr);
  project2(rh.x, rh.y, rh.n2);

  // ---- msum = project(mobius_add(dst_h, rel_h)) ----
  float dr = wsum(dh.x * rh.x + dh.y * rh.y);                 // reduction 6
  M2 ms = mobius_add2(dh.x, dh.y, dh.n2, rh.x, rh.y, rh.n2, dr);
  project2(ms.x, ms.y, ms.n2);

  // ---- tan = logmap(msum, p) = 2/lam * atanh(min(|s|,1-eps)) * s/|s| ----
  float pm = wsum(px * ms.x + py * ms.y);                     // reduction 7
  M2 sdir = mobius_add2(-px, -py, p2, ms.x, ms.y, ms.n2, -pm);
  float sn = sqrtf(fmaxf(sdir.n2, MINV));
  float z  = fminf(sn, 1.0f - EPSV);
  float at = 0.5f * logf((1.0f + z) / (1.0f - z));            // atanh
  float coef = (2.0f / lam_p) * at / sn;

  float* o = sum_kg + (size_t)s * DEMB + d0;
  atomicAdd(o,     coef * sdir.x);
  atomicAdd(o + 1, coef * sdir.y);
  if (lane == 0) atomicAdd(cnt_kg + s, 1.0f);
}

// ---------------- WMMA gate + fusion kernel ----------------
// Loads a 16x32 f16 A/B fragment per the CDNA5 ISA layout:
// lanes 0-15 vs 16-31 split K into {0-7,16-23} vs {8-15,24-31}.
__device__ __forceinline__ v16h load_frag(const float* __restrict__ row,
                                          int k0, int lane) {
  int khalf = (lane & 16) ? 8 : 0;
  v16h a;
#pragma unroll
  for (int e = 0; e < 16; ++e) {
    int k = k0 + ((e & 8) << 1) + khalf + (e & 7);
    a[e] = (_Float16)row[k];
  }
  return a;
}

__global__ void gate_fusion_kernel(const float* __restrict__ ent,
                                   const float* __restrict__ cf,
                                   const float* __restrict__ w1,
                                   const float* __restrict__ w2,
                                   float* __restrict__ fusion) {
  const int lane = threadIdx.x & 31;
  const int wid  = threadIdx.x >> 5;
  const int tile = blockIdx.x * (blockDim.x >> 5) + wid;
  if (tile * 16 >= N_ITEMS) return;       // wave-uniform; EXEC stays all-1s
  const int m0   = tile * 16;
  const int mrow = m0 + (lane & 15);

  // A fragments (rows of entity_embed / item_cf_embed), hoisted over N-tiles
  v16h a1lo = load_frag(ent + (size_t)mrow * DEMB, 0,  lane);
  v16h a1hi = load_frag(ent + (size_t)mrow * DEMB, 32, lane);
  v16h a2lo = load_frag(cf  + (size_t)mrow * DEMB, 0,  lane);
  v16h a2hi = load_frag(cf  + (size_t)mrow * DEMB, 32, lane);

#pragma unroll
  for (int n0 = 0; n0 < DEMB; n0 += 16) {
    const int nrow = n0 + (lane & 15);
    // B[k][n] = W[n][k]  (X @ W^T), so B column n = row n of W
    v16h b1lo = load_frag(w1 + (size_t)nrow * DEMB, 0,  lane);
    v16h b1hi = load_frag(w1 + (size_t)nrow * DEMB, 32, lane);
    v16h b2lo = load_frag(w2 + (size_t)nrow * DEMB, 0,  lane);
    v16h b2hi = load_frag(w2 + (size_t)nrow * DEMB, 32, lane);

    v8f c = {};
    c = __builtin_amdgcn_wmma_f32_16x16x32_f16(false, a1lo, false, b1lo, (short)0, c, false, false);
    c = __builtin_amdgcn_wmma_f32_16x16x32_f16(false, a1hi, false, b1hi, (short)0, c, false, false);
    c = __builtin_amdgcn_wmma_f32_16x16x32_f16(false, a2lo, false, b2lo, (short)0, c, false, false);
    c = __builtin_amdgcn_wmma_f32_16x16x32_f16(false, a2hi, false, b2hi, (short)0, c, false, false);

    // D layout: VGPR r, lanes 0-15 -> M=r, lanes 16-31 -> M=r+8; N = lane%16
    const int n  = n0 + (lane & 15);
    const int mb = m0 + ((lane >> 4) << 3);
#pragma unroll
    for (int r = 0; r < 8; ++r) {
      const int m = mb + r;
      float g  = 1.0f / (1.0f + __expf(-c[r]));
      float e0 = ent[(size_t)m * DEMB + n];
      float i0 = cf [(size_t)m * DEMB + n];
      fusion[(size_t)m * DEMB + n] = g * e0 + (1.0f - g) * i0;
    }
  }
}

// ---------------- bipartite UI mean-agg edge kernel (1 wave / edge) -------
__global__ void ui_edge_kernel(const float* __restrict__ fusion,
                               const float* __restrict__ usr,
                               const int* __restrict__ e_item,
                               const int* __restrict__ e_user,
                               float* __restrict__ sum_usr,
                               float* __restrict__ cnt_usr,
                               float* __restrict__ sum_it,
                               float* __restrict__ cnt_it,
                               int n_edges) {
  int gid  = blockIdx.x * blockDim.x + threadIdx.x;
  int wave = gid >> 5;
  int lane = threadIdx.x & 31;
  if (wave >= n_edges) return;

  int it = e_item[wave];
  int us = e_user[wave];
  int d0 = lane * 2;

  float2 f = *(const float2*)(fusion + (size_t)it * DEMB + d0);
  float2 u = *(const float2*)(usr    + (size_t)us * DEMB + d0);

  float* su = sum_usr + (size_t)us * DEMB + d0;
  atomicAdd(su,     f.x);
  atomicAdd(su + 1, f.y);
  float* si = sum_it + (size_t)it * DEMB + d0;
  atomicAdd(si,     u.x);
  atomicAdd(si + 1, u.y);
  if (lane == 0) {
    atomicAdd(cnt_usr + us, 1.0f);
    atomicAdd(cnt_it  + it, 1.0f);
  }
}

// ---------------- mean finalize: out[i] = sum[i] / max(cnt[i/64], 1) ------
__global__ void mean_div_kernel(const float* __restrict__ sum,
                                const float* __restrict__ cnt,
                                float* __restrict__ out, int n) {
  int i = blockIdx.x * blockDim.x + threadIdx.x;
  if (i >= n) return;
  out[i] = sum[i] / fmaxf(cnt[i >> 6], 1.0f);
}

// ---------------- host launcher ----------------
extern "C" void kernel_launch(void* const* d_in, const int* in_sizes, int n_in,
                              void* d_out, int out_size, void* d_ws, size_t ws_size,
                              hipStream_t stream) {
  (void)n_in; (void)out_size; (void)ws_size;

  const float* ent   = (const float*)d_in[0];
  const float* usr   = (const float*)d_in[1];
  const float* relt  = (const float*)d_in[2];
  const float* icf   = (const float*)d_in[3];
  const float* g1    = (const float*)d_in[4];
  const float* g2    = (const float*)d_in[5];
  const int*   sgsrc = (const int*)d_in[6];
  const int*   sgdst = (const int*)d_in[7];
  const int*   sgtyp = (const int*)d_in[8];
  const int*   ui_it = (const int*)d_in[9];
  const int*   ui_us = (const int*)d_in[10];
  const int n_kg = in_sizes[6];
  const int n_ui = in_sizes[9];

  // workspace layout (floats); zero-region first, fusion (fully overwritten) last
  float* ws      = (float*)d_ws;
  float* sum_kg  = ws;
  float* cnt_kg  = sum_kg  + (size_t)N_ENT   * DEMB;
  float* sum_usr = cnt_kg  + N_ENT;
  float* cnt_usr = sum_usr + (size_t)N_USERS * DEMB;
  float* sum_it  = cnt_usr + N_USERS;
  float* cnt_it  = sum_it  + (size_t)N_ITEMS * DEMB;
  float* fusion  = cnt_it  + N_ITEMS;
  size_t zero_floats = (size_t)(fusion - ws);
  hipMemsetAsync(d_ws, 0, zero_floats * sizeof(float), stream);

  float* out    = (float*)d_out;
  float* out_kg = out;
  float* out_u  = out_kg + (size_t)N_ENT   * DEMB;
  float* out_i  = out_u  + (size_t)N_USERS * DEMB;

  // 1) KG hyperbolic edges: one wave32 per edge
  {
    int blocks = (int)(((long long)n_kg * 32 + 255) / 256);
    kg_edge_kernel<<<blocks, 256, 0, stream>>>(ent, relt, sgsrc, sgdst, sgtyp,
                                               sum_kg, cnt_kg, n_kg);
  }

  // 2) WMMA gate + fusion: one wave per 16-row tile, 4 waves per block
  {
    int tiles  = N_ITEMS / 16;            // 1250
    int blocks = (tiles + 3) / 4;
    gate_fusion_kernel<<<blocks, 128, 0, stream>>>(ent, icf, g1, g2, fusion);
  }

  // 3) bipartite UI edges: one wave32 per edge (needs fusion)
  {
    int blocks = (int)(((long long)n_ui * 32 + 255) / 256);
    ui_edge_kernel<<<blocks, 256, 0, stream>>>(fusion, usr, ui_it, ui_us,
                                               sum_usr, cnt_usr, sum_it, cnt_it,
                                               n_ui);
  }

  // 4) finalize means into the packed output (out, u, i_cf)
  {
    int n0 = N_ENT * DEMB;
    mean_div_kernel<<<(n0 + 255) / 256, 256, 0, stream>>>(sum_kg, cnt_kg, out_kg, n0);
    int n1 = N_USERS * DEMB;
    mean_div_kernel<<<(n1 + 255) / 256, 256, 0, stream>>>(sum_usr, cnt_usr, out_u, n1);
    int n2 = N_ITEMS * DEMB;
    mean_div_kernel<<<(n2 + 255) / 256, 256, 0, stream>>>(sum_it, cnt_it, out_i, n2);
  }
}
